// GCN_LSTM_83743272337792
// MI455X (gfx1250) — matrix operand order repaired and checked by reading, not compiled, not run on hardware
//
#include <hip/hip_runtime.h>
#include <hip/hip_bf16.h>
#include <math.h>

typedef __attribute__((ext_vector_type(16))) __bf16 v16bf;
typedef __attribute__((ext_vector_type(8)))  float  v8f;

#define H 128
#define FOURH 512

// ---------------------------------------------------------------------------
// Generic utility kernels
// ---------------------------------------------------------------------------
__global__ void zero_f32(float* p, long long n) {
    long long i = (long long)blockIdx.x * blockDim.x + threadIdx.x;
    if (i < n) p[i] = 0.0f;
}

// ---------------------------------------------------------------------------
// Degree / symmetric normalization
// ---------------------------------------------------------------------------
__global__ void deg_count(const int* ei, float* deg, int E) {
    int e = blockIdx.x * blockDim.x + threadIdx.x;
    if (e < E) atomicAdd(&deg[ei[E + e]], 1.0f);
}

__global__ void deg_finalize(float* deg, int n) {
    int i = blockIdx.x * blockDim.x + threadIdx.x;
    if (i < n) {
        float d = deg[i] + 1.0f;           // self loop
        deg[i] = rsqrtf(d);                // in-place -> dinv
    }
}

// ---------------------------------------------------------------------------
// Pack B (f32, [K,Nc] or [Nc,K]) into bf16 WMMA B-fragment layout.
// Chunk order is [kc][ct] so that, for a fixed k-chunk, the NT adjacent
// column-tile fragments are contiguous (immediate-offset loads in the GEMM).
//   chunk = kc*ntN + ct
//   Bp[(chunk*32 + lane)*16 + i] = B[k][ct*16 + (lane&15)],
//   k = kc*32 + (lane>>4)*16 + i            (contiguous k order, same as load_A_frag)
// ---------------------------------------------------------------------------
__global__ void pack_B_frag(const float* __restrict__ B, __bf16* __restrict__ Bp,
                            int Nc, int K, int transB, int ldb) {
    int idx = blockIdx.x * blockDim.x + threadIdx.x;
    int nkc = K >> 5;
    int ntN = Nc >> 4;
    int total = ntN * nkc * 32;
    if (idx >= total) return;
    int lane  = idx & 31;
    int chunk = idx >> 5;
    int ct = chunk % ntN;
    int kc = chunk / ntN;
    int c  = lane & 15;
    int hi = lane >> 4;
    int col = ct * 16 + c;
    __bf16* dst = Bp + (size_t)idx * 16;
    #pragma unroll
    for (int i = 0; i < 16; ++i) {
        int k = kc * 32 + hi * 16 + i;
        float v = transB ? B[(size_t)col * ldb + k] : B[(size_t)k * ldb + col];
        dst[i] = (__bf16)v;
    }
}

// ---------------------------------------------------------------------------
// A fragment loader (f32 -> bf16 on the fly), 16x32 tile, lane layout:
// lanes 0-15 hold K 0..15, lanes 16-31 hold K 16..31.
// ---------------------------------------------------------------------------
__device__ inline v16bf load_A_frag(const float* A, int lda, int rowBase,
                                    int kBase, int lane) {
    int r  = lane & 15;
    int hi = lane >> 4;
    const float* ap = A + (size_t)(rowBase + r) * lda + kBase + hi * 16;
    float4 x0 = *(const float4*)(ap + 0);
    float4 x1 = *(const float4*)(ap + 4);
    float4 x2 = *(const float4*)(ap + 8);
    float4 x3 = *(const float4*)(ap + 12);
    v16bf a;
    a[0]=(__bf16)x0.x; a[1]=(__bf16)x0.y; a[2]=(__bf16)x0.z; a[3]=(__bf16)x0.w;
    a[4]=(__bf16)x1.x; a[5]=(__bf16)x1.y; a[6]=(__bf16)x1.z; a[7]=(__bf16)x1.w;
    a[8]=(__bf16)x2.x; a[9]=(__bf16)x2.y; a[10]=(__bf16)x2.z; a[11]=(__bf16)x2.w;
    a[12]=(__bf16)x3.x; a[13]=(__bf16)x3.y; a[14]=(__bf16)x3.z; a[15]=(__bf16)x3.w;
    return a;
}

// ---------------------------------------------------------------------------
// WMMA bf16 GEMM: C[M,Nc] = A[M,K] @ B (+bias(+bias2)), B pre-packed.
// Each wave computes NT adjacent 16x16 column tiles for one 16-row stripe,
// reusing the A fragment across NT v_wmma issues.
// ---------------------------------------------------------------------------
template <int NT>
__global__ __launch_bounds__(256)
void gemm_wmma_bf16(const float* __restrict__ A, const __bf16* __restrict__ Bp,
                    const float* __restrict__ bias, const float* __restrict__ bias2,
                    float* __restrict__ C,
                    int M, int Nc, int K, int lda, int ldc) {
    int lane    = threadIdx.x & 31;
    int waveId  = blockIdx.x * (blockDim.x >> 5) + (threadIdx.x >> 5);
    int ntN     = Nc >> 4;
    int nGroups = ntN / NT;
    int tiles   = (M >> 4) * nGroups;
    if (waveId >= tiles) return;

    int rowBase = (waveId / nGroups) * 16;
    int cg      = (waveId % nGroups) * NT;      // first column tile of group
    int nkc     = K >> 5;

    // per-lane base: fragments for (kc, cg+t) live at bbase + kc*ntN*512 + t*512
    const __bf16* bbase = Bp + (size_t)lane * 16 + (size_t)cg * 512;
    const size_t  kstep = (size_t)ntN * 512;

    v8f acc[NT];
    #pragma unroll
    for (int t = 0; t < NT; ++t) acc[t] = (v8f){};

    #pragma unroll 4
    for (int kc = 0; kc < nkc; ++kc) {
        v16bf af = load_A_frag(A, lda, rowBase, kc * 32, lane);
        const __bf16* bp = bbase + (size_t)kc * kstep;
        #pragma unroll
        for (int t = 0; t < NT; ++t) {
            v16bf bfr = *(const v16bf*)(bp + (size_t)t * 512);
            acc[t] = __builtin_amdgcn_wmma_f32_16x16x32_bf16(
                false, af, false, bfr, (short)0, acc[t], false, false);
        }
    }

    int c  = lane & 15;
    int hi = lane >> 4;
    #pragma unroll
    for (int t = 0; t < NT; ++t) {
        int col = (cg + t) * 16 + c;
        float badd = 0.0f;
        if (bias)  badd += bias[col];
        if (bias2) badd += bias2[col];
        #pragma unroll
        for (int v = 0; v < 8; ++v) {
            int row = rowBase + v + hi * 8;
            C[(size_t)row * ldc + col] = acc[t][v] + badd;
        }
    }
}

// ---------------------------------------------------------------------------
// GCN neighbor aggregation (wave per edge) + self-loop/bias/relu finalize
// ---------------------------------------------------------------------------
__global__ __launch_bounds__(256)
void gcn_scatter(const int* __restrict__ ei, const float* __restrict__ xw,
                 const float* __restrict__ dinv, float* __restrict__ out, int E) {
    int wave = blockIdx.x * (blockDim.x >> 5) + (threadIdx.x >> 5);
    if (wave >= E) return;
    int lane = threadIdx.x & 31;
    int s = ei[wave];
    int d = ei[E + wave];
    float nrm = dinv[s] * dinv[d];
    const float* xs = xw + (size_t)s * H;
    float* od = out + (size_t)d * H;
    #pragma unroll
    for (int r = 0; r < 4; ++r) {
        int j = lane + r * 32;
        atomicAdd(&od[j], xs[j] * nrm);
    }
}

__global__ void gcn_finalize(const float* __restrict__ xw,
                             const float* __restrict__ dinv,
                             const float* __restrict__ b,
                             float* __restrict__ out, long long n, int relu) {
    long long idx = (long long)blockIdx.x * blockDim.x + threadIdx.x;
    if (idx >= n) return;
    int i = (int)(idx >> 7);
    int j = (int)(idx & 127);
    float di = dinv[i];
    float v = out[idx] + xw[idx] * di * di + b[j];
    if (relu) v = fmaxf(v, 0.0f);
    out[idx] = v;
}

// ---------------------------------------------------------------------------
// Async global->LDS copy (CDNA5), tracked by ASYNCcnt.
// ---------------------------------------------------------------------------
__device__ inline void async_load_f32_to_lds(const float* gaddr, unsigned ldsByteOff) {
    asm volatile("global_load_async_to_lds_b32 %0, %1, off"
                 :: "v"(ldsByteOff), "v"(gaddr) : "memory");
}

// ---------------------------------------------------------------------------
// Persistent single-workgroup LSTM recurrence. whh^T resident in LDS (bf16).
// gx already contains seq@wih^T + (bih+bhh). Each step's 512-float gate row
// is double-buffered in LDS via global_load_async_to_lds_b32, issued one
// step ahead and consumed after s_wait_asynccnt.
// ---------------------------------------------------------------------------
__device__ inline float sigmoidf_(float x) { return 1.0f / (1.0f + expf(-x)); }

__global__ __launch_bounds__(512)
void lstm_recurrence(const float* __restrict__ gx, const float* __restrict__ whh,
                     float* __restrict__ hs_out,
                     float* __restrict__ hT_out, float* __restrict__ cT_out, int T) {
    extern __shared__ char smem[];
    __bf16* wT    = (__bf16*)smem;                         // [128][512]
    float*  hbuf  = (float*)(smem + (size_t)H * FOURH * 2);
    float*  cbuf  = hbuf + H;
    float*  gbuf  = cbuf + H;                              // [512]
    float*  gxlds = gbuf + FOURH;                          // [2][512] dbl buffer
    int tid = threadIdx.x;

    // stage whh^T into LDS: wT[k*512 + n] = whh[n*128 + k]
    for (int idx = tid; idx < FOURH * H; idx += FOURH) {
        int n = idx >> 7;
        int k = idx & 127;
        wT[k * FOURH + n] = (__bf16)whh[idx];
    }
    if (tid < H) { hbuf[tid] = 0.0f; cbuf[tid] = 0.0f; }

    // preload step 0 gate row into buffer 0 (per-lane async copy)
    unsigned myOff0 = (unsigned)(size_t)(gxlds + tid);
    unsigned myOff1 = (unsigned)(size_t)(gxlds + FOURH + tid);
    async_load_f32_to_lds(gx + tid, myOff0);
    __syncthreads();

    for (int t = 0; t < T; ++t) {
        if (t + 1 < T) {
            // issue next step's row into the other buffer, then wait for the
            // current one (per-wave ASYNCcnt is in-order: <=1 means t's done)
            async_load_f32_to_lds(gx + (size_t)(t + 1) * FOURH + tid,
                                  ((t + 1) & 1) ? myOff1 : myOff0);
            asm volatile("s_wait_asynccnt 0x1" ::: "memory");
        } else {
            asm volatile("s_wait_asynccnt 0x0" ::: "memory");
        }
        float acc = gxlds[(t & 1) * FOURH + tid];
        const __bf16* wcol = wT + tid;
        #pragma unroll 8
        for (int k = 0; k < H; ++k)
            acc += hbuf[k] * (float)wcol[(size_t)k * FOURH];
        gbuf[tid] = acc;
        __syncthreads();
        if (tid < H) {
            float gi = sigmoidf_(gbuf[tid]);
            float gf = sigmoidf_(gbuf[H + tid]);
            float gg = tanhf(gbuf[2 * H + tid]);
            float go = sigmoidf_(gbuf[3 * H + tid]);
            float c = gf * cbuf[tid] + gi * gg;
            float h = go * tanhf(c);
            cbuf[tid] = c;
            hbuf[tid] = h;
            hs_out[(size_t)t * H + tid] = h;
        }
        __syncthreads();
    }
    if (tid < H) {
        hT_out[tid] = hbuf[tid];
        cT_out[tid] = cbuf[tid];
    }
}

// ---------------------------------------------------------------------------
// Edge head constants: eWc[k] = sum_j eW[k][j]*cW3[j];  ebc = cb + eb.cW3
// ---------------------------------------------------------------------------
__global__ void edge_consts(const float* __restrict__ eW, const float* __restrict__ eb,
                            const float* __restrict__ cW, const float* __restrict__ cb,
                            float* __restrict__ eWc, float* __restrict__ ebc) {
    int t = threadIdx.x;
    if (t < 32) {
        float s = 0.0f;
        for (int j = 0; j < H; ++j) s += eW[t * H + j] * cW[2 * H + j];
        eWc[t] = s;
    } else if (t == 32) {
        float s = cb[0];
        for (int j = 0; j < H; ++j) s += eb[j] * cW[2 * H + j];
        ebc[0] = s;
    }
}

// ---------------------------------------------------------------------------
// Edge head (wave per edge): gather src/dst rows (also outputs), fused dots.
// ---------------------------------------------------------------------------
__global__ __launch_bounds__(256)
void edge_final(const int* __restrict__ ei, const float* __restrict__ lstm,
                const float* __restrict__ ef, const float* __restrict__ cW,
                const float* __restrict__ eWc, const float* __restrict__ ebc,
                float* __restrict__ pred, float* __restrict__ osrc,
                float* __restrict__ odst, int E) {
    int wave = blockIdx.x * (blockDim.x >> 5) + (threadIdx.x >> 5);
    if (wave >= E) return;
    int lane = threadIdx.x & 31;
    int s = ei[wave];
    int d = ei[E + wave];
    const float* sv = lstm + (size_t)s * H;
    const float* dv = lstm + (size_t)d * H;
    float acc = 0.0f;
    #pragma unroll
    for (int r = 0; r < 4; ++r) {
        int j = lane + r * 32;
        float a = sv[j];
        float b = dv[j];
        osrc[(size_t)wave * H + j] = a;
        odst[(size_t)wave * H + j] = b;
        acc += a * cW[j] + b * cW[H + j];
    }
    acc += ef[(size_t)wave * 32 + lane] * eWc[lane];
    #pragma unroll
    for (int off = 16; off > 0; off >>= 1) acc += __shfl_xor(acc, off);
    if (lane == 0) pred[wave] = acc + ebc[0];
}

// ---------------------------------------------------------------------------
// Host side
// ---------------------------------------------------------------------------
extern "C" void kernel_launch(void* const* d_in, const int* in_sizes, int n_in,
                              void* d_out, int out_size, void* d_ws, size_t ws_size,
                              hipStream_t stream) {
    const float* x    = (const float*)d_in[0];
    const int*   ei   = (const int*)d_in[1];
    const float* ef   = (const float*)d_in[2];
    const float* W1   = (const float*)d_in[3];
    const float* b1   = (const float*)d_in[4];
    const float* W2   = (const float*)d_in[5];
    const float* b2   = (const float*)d_in[6];
    const float* wih0 = (const float*)d_in[7];
    const float* whh0 = (const float*)d_in[8];
    const float* bih0 = (const float*)d_in[9];
    const float* bhh0 = (const float*)d_in[10];
    const float* wih1 = (const float*)d_in[11];
    const float* whh1 = (const float*)d_in[12];
    const float* bih1 = (const float*)d_in[13];
    const float* bhh1 = (const float*)d_in[14];
    const float* eW   = (const float*)d_in[15];
    const float* eb   = (const float*)d_in[16];
    const float* cW   = (const float*)d_in[17];
    const float* cb   = (const float*)d_in[18];

    const int Nn = in_sizes[0] / H;    // 20000
    const int Ee = in_sizes[1] / 2;    // 320000

    float* ws = (float*)d_ws;
    const size_t nf = (size_t)Nn * H;
    float* xw   = ws;                  // [N,128]  staging for both convs
    float* h1   = ws + 1 * nf;         // [N,128]  conv1 out (relu)
    float* h2   = ws + 2 * nf;         // [N,128]  conv2 out (LSTM input)
    float* hs0  = ws + 3 * nf;         // [N,128]  LSTM layer0 hidden seq
    float* hs1  = ws + 4 * nf;         // [N,128]  LSTM layer1 hidden seq
    float* gx   = ws + 5 * nf;         // [N,512]  gate precompute (reused)
    float* dinv = ws + 9 * nf;         // [N]      deg -> dinv in place
    float* eWc  = dinv + Nn;           // [32]
    float* ebc  = eWc + 32;            // [1]
    size_t bpOff = ((9 * nf + (size_t)Nn + 33 + 63) / 64) * 64;   // 256B aligned
    __bf16* Bpack = (__bf16*)(ws + bpOff);   // up to 128 KB fragment-packed B

    float* out  = (float*)d_out;
    float* pred = out;                 // [E]
    float* hn   = out + Ee;            // [2,128]
    float* cn   = hn + 2 * H;          // [2,128]
    float* osrc = cn + 2 * H;          // [E,128]
    float* odst = osrc + (size_t)Ee * H;

    auto cdiv = [](long long a, long long b) { return (int)((a + b - 1) / b); };
    const int waveBlocks = cdiv(Ee, 8);        // 8 waves / 256-thread block

    // ---- degrees / normalization ----
    zero_f32<<<cdiv(Nn, 256), 256, 0, stream>>>(dinv, Nn);
    deg_count<<<cdiv(Ee, 256), 256, 0, stream>>>(ei, dinv, Ee);
    deg_finalize<<<cdiv(Nn, 256), 256, 0, stream>>>(dinv, Nn);

    // pack-kernel thread counts
    const int packH  = (H / 16) * (H / 32) * 32;        // Nc=128, K=128
    const int pack4H = (FOURH / 16) * (H / 32) * 32;    // Nc=512, K=128

    // ---- GCN conv 1 ----
    pack_B_frag<<<cdiv(packH, 256), 256, 0, stream>>>(W1, Bpack, H, H, 0, H);
    {
        int tiles = (Nn / 16) * ((H / 16) / 4);
        gemm_wmma_bf16<4><<<cdiv(tiles, 8), 256, 0, stream>>>(
            x, Bpack, nullptr, nullptr, xw, Nn, H, H, H, H);
    }
    zero_f32<<<cdiv(nf, 256), 256, 0, stream>>>(h1, (long long)nf);
    gcn_scatter<<<waveBlocks, 256, 0, stream>>>(ei, xw, dinv, h1, Ee);
    gcn_finalize<<<cdiv(nf, 256), 256, 0, stream>>>(xw, dinv, b1, h1, (long long)nf, 1);

    // ---- GCN conv 2 ----
    pack_B_frag<<<cdiv(packH, 256), 256, 0, stream>>>(W2, Bpack, H, H, 0, H);
    {
        int tiles = (Nn / 16) * ((H / 16) / 4);
        gemm_wmma_bf16<4><<<cdiv(tiles, 8), 256, 0, stream>>>(
            h1, Bpack, nullptr, nullptr, xw, Nn, H, H, H, H);
    }
    zero_f32<<<cdiv(nf, 256), 256, 0, stream>>>(h2, (long long)nf);
    gcn_scatter<<<waveBlocks, 256, 0, stream>>>(ei, xw, dinv, h2, Ee);
    gcn_finalize<<<cdiv(nf, 256), 256, 0, stream>>>(xw, dinv, b2, h2, (long long)nf, 0);

    const size_t lstmSmem = (size_t)H * FOURH * 2 +
                            (H + H + FOURH + 2 * FOURH) * sizeof(float);

    // ---- LSTM layer 0 ----
    pack_B_frag<<<cdiv(pack4H, 256), 256, 0, stream>>>(wih0, Bpack, FOURH, H, 1, H);
    {
        int tiles = (Nn / 16) * ((FOURH / 16) / 4);
        gemm_wmma_bf16<4><<<cdiv(tiles, 8), 256, 0, stream>>>(
            h2, Bpack, bih0, bhh0, gx, Nn, FOURH, H, H, FOURH);
    }
    lstm_recurrence<<<1, 512, lstmSmem, stream>>>(gx, whh0, hs0, hn, cn, Nn);

    // ---- LSTM layer 1 ----
    pack_B_frag<<<cdiv(pack4H, 256), 256, 0, stream>>>(wih1, Bpack, FOURH, H, 1, H);
    {
        int tiles = (Nn / 16) * ((FOURH / 16) / 4);
        gemm_wmma_bf16<4><<<cdiv(tiles, 8), 256, 0, stream>>>(
            hs0, Bpack, bih1, bhh1, gx, Nn, FOURH, H, H, FOURH);
    }
    lstm_recurrence<<<1, 512, lstmSmem, stream>>>(gx, whh1, hs1, hn + H, cn + H, Nn);

    // ---- edge head ----
    edge_consts<<<1, 64, 0, stream>>>(eW, eb, cW, cb, eWc, ebc);
    edge_final<<<waveBlocks, 256, 0, stream>>>(
        ei, hs1, ef, cW, eWc, ebc, pred, osrc, odst, Ee);
}
